// MultiheadAttention_50594714747419
// MI455X (gfx1250) — matrix operand order repaired and checked by reading, not compiled, not run on hardware
//
#include <hip/hip_runtime.h>
#include <hip/hip_bf16.h>

// ---------------------------------------------------------------------------
// MHA forward on gfx1250: bf16 WMMA (16x16x32) everywhere, fp32 accumulate.
// ---------------------------------------------------------------------------

typedef __attribute__((ext_vector_type(16))) __bf16 bf16x16;
typedef __attribute__((ext_vector_type(8)))  float  f32x8;

#define EMBED 1024
#define SEQ   2048
#define BATCH 2
#define HEADS 16
#define HDIM  64
#define MTOT  (BATCH * SEQ)   // 4096 rows for all projection GEMMs

static __device__ __forceinline__ unsigned short f2bf(float f) {
  union { __bf16 h; unsigned short u; } cv;
  cv.h = (__bf16)f;
  return cv.u;
}

static __device__ __forceinline__ f32x8 wmma_bf16(bf16x16 a, bf16x16 b, f32x8 c) {
  // D = A(16x32 bf16) * B(32x16 bf16) + C(16x16 f32)
  return __builtin_amdgcn_wmma_f32_16x16x32_bf16(
      /*neg_a=*/false, a, /*neg_b=*/false, b,
      /*c_mod=*/(short)0, c, /*reuse_a=*/false, /*reuse_b=*/false);
}

union FragU { bf16x16 v; uint4 q[2]; };

// ---------------------------------------------------------------------------
// fp32 -> bf16 elementwise convert (vectorized x4)
// ---------------------------------------------------------------------------
__global__ __launch_bounds__(256) void cvt_f32_bf16(
    const float* __restrict__ s, unsigned short* __restrict__ d, int n4) {
  int i = blockIdx.x * 256 + threadIdx.x;
  if (i < n4) {
    float4 f = ((const float4*)s)[i];
    ushort4 h;
    h.x = f2bf(f.x); h.y = f2bf(f.y); h.z = f2bf(f.z); h.w = f2bf(f.w);
    ((ushort4*)d)[i] = h;
  }
}

// ---------------------------------------------------------------------------
// Tiled WMMA Linear: y[m][n] = sum_k A[m][k] * W[n][k] + bias[n]
// Block tile 128(M) x 64(N), BK=32, double-buffered LDS, 256 threads = 8 waves
// arranged 4(M) x 2(N); each wave computes 32x32 (2 A-frags x 2 B-frags,
// 4 WMMA per K-step from 8 LDS b128 loads).
// MODE 0: bf16 out, layout [b][h][s][d]          (Q, K)
// MODE 1: bf16 out, layout [b][h][d][s]          (V transposed for P*V)
// MODE 2: f32  out, row-major [m][n]             (final projection)
// ---------------------------------------------------------------------------
template <int MODE>
__global__ __launch_bounds__(256) void linear_wmma(
    const unsigned short* __restrict__ A,   // [MTOT][EMBED] bf16
    const unsigned short* __restrict__ W,   // [EMBED][EMBED] bf16 (row n = B col)
    const float* __restrict__ bias,         // [EMBED] f32
    void* __restrict__ out) {
  constexpr int K = EMBED;
  constexpr int N = EMBED;
  constexpr int BM = 128, BN = 64, BK = 32;
  constexpr int LDT = 40;                    // padded row stride (bf16 elems)

  __shared__ unsigned short As[2][BM * LDT]; // 2 x 10240 B
  __shared__ unsigned short Bs[2][BN * LDT]; // 2 x  5120 B

  const int m0 = blockIdx.x * BM;
  const int n0 = blockIdx.y * BN;
  const int tid  = threadIdx.x;
  const int wave = tid >> 5;
  const int lane = tid & 31;
  const int lr   = lane & 15;
  const bool hi  = lane >= 16;
  const int mw = (wave & 3) * 32;            // wave M offset in tile
  const int nw = (wave >> 2) * 32;           // wave N offset in tile

  f32x8 acc[2][2];
#pragma unroll
  for (int i = 0; i < 2; ++i)
#pragma unroll
    for (int j = 0; j < 2; ++j)
#pragma unroll
      for (int e = 0; e < 8; ++e) acc[i][j][e] = 0.0f;

  const int lrow = tid >> 2;         // 0..63
  const int lcol = (tid & 3) * 8;    // bf16-element offset (16B chunks)
  const unsigned short* gA0 = A + (m0 + lrow) * K + lcol;
  const unsigned short* gA1 = A + (m0 + lrow + 64) * K + lcol;
  const unsigned short* gB  = W + (n0 + lrow) * K + lcol;

  // Prologue: stage K-tile 0 into buffer 0
  {
    uint4 ra0 = *(const uint4*)gA0;
    uint4 ra1 = *(const uint4*)gA1;
    uint4 rb  = *(const uint4*)gB;
    *(uint4*)(As[0] + lrow * LDT + lcol)        = ra0;
    *(uint4*)(As[0] + (lrow + 64) * LDT + lcol) = ra1;
    *(uint4*)(Bs[0] + lrow * LDT + lcol)        = rb;
  }

  for (int kt = 0; kt < K; kt += BK) {
    const int cur = (kt >> 5) & 1;
    __syncthreads();   // staged tile visible; prior reads of other buffer done

    // Issue next tile's global loads early (overlap with WMMA below)
    const bool more = (kt + BK) < K;
    uint4 na0, na1, nb;
    if (more) {
      na0 = *(const uint4*)(gA0 + kt + BK);
      na1 = *(const uint4*)(gA1 + kt + BK);
      nb  = *(const uint4*)(gB  + kt + BK);
      if (kt + 2 * BK < K) {   // global_prefetch_b8 two tiles ahead
        __builtin_prefetch(gA0 + kt + 2 * BK, 0, 0);
        __builtin_prefetch(gA1 + kt + 2 * BK, 0, 0);
        __builtin_prefetch(gB  + kt + 2 * BK, 0, 0);
      }
    }

    // A fragments 16x32 (ISA layout): lanes 0-15 K{0..7,16..23}, 16-31 K{8..15,24..31}
    FragU a[2];
#pragma unroll
    for (int i = 0; i < 2; ++i) {
      const unsigned short* ap =
          As[cur] + (mw + i * 16 + lr) * LDT + (hi ? 8 : 0);
      a[i].q[0] = *(const uint4*)ap;
      a[i].q[1] = *(const uint4*)(ap + 16);
    }
#pragma unroll
    for (int j = 0; j < 2; ++j) {
      // B fragment 32x16: lane col = lr, lanes 0-15 K{0..7,16..23}, 16-31 K{8..15,24..31}
      FragU b;
      const unsigned short* bp =
          Bs[cur] + (nw + j * 16 + lr) * LDT + (hi ? 8 : 0);
      b.q[0] = *(const uint4*)bp;
      b.q[1] = *(const uint4*)(bp + 16);
#pragma unroll
      for (int i = 0; i < 2; ++i)
        acc[i][j] = wmma_bf16(a[i].v, b.v, acc[i][j]);
    }

    // Stage next tile into the other buffer (safe: top barrier of this
    // iteration guaranteed everyone finished reading it)
    if (more) {
      const int nxt = 1 - cur;
      *(uint4*)(As[nxt] + lrow * LDT + lcol)        = na0;
      *(uint4*)(As[nxt] + (lrow + 64) * LDT + lcol) = na1;
      *(uint4*)(Bs[nxt] + lrow * LDT + lcol)        = nb;
    }
  }

  // Epilogue: C layout VGPR v -> (M = v + (hi?8:0), N = lr)
#pragma unroll
  for (int j = 0; j < 2; ++j) {
    const int n  = n0 + nw + j * 16 + lr;
    const float bv = bias[n];
#pragma unroll
    for (int i = 0; i < 2; ++i) {
#pragma unroll
      for (int v = 0; v < 8; ++v) {
        const int m = m0 + mw + i * 16 + v + (hi ? 8 : 0);
        const float val = acc[i][j][v] + bv;
        if (MODE == 2) {
          ((float*)out)[m * N + n] = val;
        } else {
          const int b = m >> 11, s = m & (SEQ - 1);
          const int h = n >> 6,  d = n & 63;
          if (MODE == 0)
            ((unsigned short*)out)[(((b * HEADS + h) * SEQ + s) << 6) + d] =
                f2bf(val);
          else
            ((unsigned short*)out)[((b * HEADS + h) * HDIM + d) * SEQ + s] =
                f2bf(val);
        }
      }
    }
  }
}

// ---------------------------------------------------------------------------
// Flash attention: per block = one (b,h) and 128 s-rows; 8 waves x 16 rows.
// Q fragments in registers; K/V B-fragments loaded directly from L2-resident
// global; online softmax. Row max via 16-lane shuffles; row SUM via a WMMA
// against an all-ones B matrix (P is already in A-fragment form for P*V, and
// P x ones broadcasts each row's sum into that row's C-layout slots) --
// removes 32 ds_bpermute + waits per tile.
// ---------------------------------------------------------------------------
__global__ __launch_bounds__(256) void flash_attn_wmma(
    const unsigned short* __restrict__ Q,   // [B][H][S][D] bf16
    const unsigned short* __restrict__ Kb,  // [B][H][S][D] bf16
    const unsigned short* __restrict__ Vt,  // [B][H][D][S] bf16
    unsigned short* __restrict__ O) {       // [B*S][E] bf16
  constexpr int PLD = 40;
  __shared__ unsigned short Ps[8 * 16 * PLD];   // per-wave 16x32 P scratch

  const int sblk = blockIdx.x;                  // 0..S/128-1
  const int bh   = blockIdx.y;                  // 0..B*H-1
  const int b = bh / HEADS, h = bh % HEADS;
  const int tid  = threadIdx.x;
  const int wave = tid >> 5;
  const int lane = tid & 31;
  const int lr   = lane & 15;
  const bool hi  = lane >= 16;
  const int srow = sblk * 128 + wave * 16;      // wave's first s-row

  // Load this wave's Q rows as two 16x32 A-fragments (K = d)
  const unsigned short* qbase = Q + ((size_t)bh * SEQ + (srow + lr)) * HDIM;
  FragU qf[2];
#pragma unroll
  for (int kk = 0; kk < 2; ++kk) {
    const unsigned short* qp = qbase + kk * 32 + (hi ? 8 : 0);
    qf[kk].q[0] = *(const uint4*)qp;
    qf[kk].q[1] = *(const uint4*)(qp + 16);
  }

  // All-ones bf16 B fragment (0x3F80 per element) for row-sum WMMA
  FragU ones;
#pragma unroll
  for (int e = 0; e < 2; ++e) {
    ones.q[e].x = 0x3F803F80u; ones.q[e].y = 0x3F803F80u;
    ones.q[e].z = 0x3F803F80u; ones.q[e].w = 0x3F803F80u;
  }

  f32x8 o[4];
  float mrow[8], lsum[8];
#pragma unroll
  for (int j = 0; j < 4; ++j)
#pragma unroll
    for (int e = 0; e < 8; ++e) o[j][e] = 0.0f;
#pragma unroll
  for (int v = 0; v < 8; ++v) { mrow[v] = -1e30f; lsum[v] = 0.0f; }

  unsigned short* pw = Ps + wave * 16 * PLD;

  for (int t0 = 0; t0 < SEQ; t0 += 32) {
    // ---- scores tile: S = Q(16x64) * K^T(64x32) -> 16x32 fp32 ----
    f32x8 sa[2];
#pragma unroll
    for (int nf = 0; nf < 2; ++nf)
#pragma unroll
      for (int e = 0; e < 8; ++e) sa[nf][e] = 0.0f;

#pragma unroll
    for (int nf = 0; nf < 2; ++nf) {
      const int t = t0 + nf * 16 + lr;
      const unsigned short* kp =
          Kb + ((size_t)bh * SEQ + t) * HDIM + (hi ? 8 : 0);
#pragma unroll
      for (int kk = 0; kk < 2; ++kk) {
        FragU kf;                                       // B frag, K = d
        kf.q[0] = *(const uint4*)(kp + kk * 32);
        kf.q[1] = *(const uint4*)(kp + kk * 32 + 16);
        sa[nf] = wmma_bf16(qf[kk].v, kf.v, sa[nf]);
      }
    }

    // ---- online softmax (max via shuffles; sum via WMMA below) ----
    float alpha[8];
#pragma unroll
    for (int v = 0; v < 8; ++v) {
      const float s0 = sa[0][v] * 0.125f;   // 1/sqrt(64)
      const float s1 = sa[1][v] * 0.125f;
      float t = fmaxf(s0, s1);
      t = fmaxf(t, __shfl_xor(t, 1, 16));
      t = fmaxf(t, __shfl_xor(t, 2, 16));
      t = fmaxf(t, __shfl_xor(t, 4, 16));
      t = fmaxf(t, __shfl_xor(t, 8, 16));
      const float mn = fmaxf(mrow[v], t);
      alpha[v] = __expf(mrow[v] - mn);
      mrow[v] = mn;
      const float p0 = __expf(s0 - mn);
      const float p1 = __expf(s1 - mn);
      // C layout -> LDS (row-major 16x32 bf16) for A-frag reload
      const int row = v + (hi ? 8 : 0);
      pw[row * PLD + lr]      = f2bf(p0);
      pw[row * PLD + 16 + lr] = f2bf(p1);
    }

    // rescale running O by alpha (per-row)
#pragma unroll
    for (int nf = 0; nf < 4; ++nf)
#pragma unroll
      for (int v = 0; v < 8; ++v) o[nf][v] *= alpha[v];

    // reload P as 16x32 A-fragment (K = t); intra-wave DS RAW -> s_wait_dscnt
    FragU pf;
    const unsigned short* pp = pw + lr * PLD + (hi ? 8 : 0);
    pf.q[0] = *(const uint4*)pp;
    pf.q[1] = *(const uint4*)(pp + 16);

    // row sums: P(16x32) x ones(32x16); every lane's slot v holds its row's sum
    f32x8 zero;
#pragma unroll
    for (int e = 0; e < 8; ++e) zero[e] = 0.0f;
    const f32x8 rowsum = wmma_bf16(pf.v, ones.v, zero);
#pragma unroll
    for (int v = 0; v < 8; ++v) lsum[v] = lsum[v] * alpha[v] + rowsum[v];

    // ---- O += P(16x32) * V(32x64) ----
#pragma unroll
    for (int nf = 0; nf < 4; ++nf) {
      const int d = nf * 16 + lr;
      const unsigned short* vp =
          Vt + ((size_t)bh * HDIM + d) * SEQ + t0 + (hi ? 8 : 0);
      FragU vf;                                         // B frag, K = t
      vf.q[0] = *(const uint4*)vp;
      vf.q[1] = *(const uint4*)(vp + 16);
      o[nf] = wmma_bf16(pf.v, vf.v, o[nf]);
    }
  }

  // ---- normalize and write attn in [b][s][h*64+d] bf16 for the out-proj ----
#pragma unroll
  for (int nf = 0; nf < 4; ++nf) {
#pragma unroll
    for (int v = 0; v < 8; ++v) {
      const int s = srow + v + (hi ? 8 : 0);
      const float val = o[nf][v] / lsum[v];
      O[((size_t)(b * SEQ + s)) * EMBED + h * HDIM + nf * 16 + lr] = f2bf(val);
    }
  }
}

// ---------------------------------------------------------------------------
// Host-side orchestration
// ---------------------------------------------------------------------------
extern "C" void kernel_launch(void* const* d_in, const int* in_sizes, int n_in,
                              void* d_out, int out_size, void* d_ws, size_t ws_size,
                              hipStream_t stream) {
  (void)in_sizes; (void)n_in; (void)out_size; (void)ws_size;
  const float* x  = (const float*)d_in[0];
  const float* wq = (const float*)d_in[1];
  const float* bq = (const float*)d_in[2];
  const float* wk = (const float*)d_in[3];
  const float* bk = (const float*)d_in[4];
  const float* wv = (const float*)d_in[5];
  const float* bv = (const float*)d_in[6];
  const float* wo = (const float*)d_in[7];
  const float* bo = (const float*)d_in[8];

  char* ws = (char*)d_ws;
  const size_t MiB = 1024 * 1024;
  unsigned short* xb  = (unsigned short*)(ws + 0);        // 8 MiB
  unsigned short* wqb = (unsigned short*)(ws + 8  * MiB); // 2 MiB each
  unsigned short* wkb = (unsigned short*)(ws + 10 * MiB);
  unsigned short* wvb = (unsigned short*)(ws + 12 * MiB);
  unsigned short* wob = (unsigned short*)(ws + 14 * MiB);
  unsigned short* qb  = (unsigned short*)(ws + 16 * MiB); // 8 MiB
  unsigned short* kb  = (unsigned short*)(ws + 24 * MiB); // 8 MiB
  unsigned short* vtb = (unsigned short*)(ws + 32 * MiB); // 8 MiB
  unsigned short* ab  = (unsigned short*)(ws + 40 * MiB); // 8 MiB

  const int nx4 = MTOT * EMBED / 4;
  const int nw4 = EMBED * EMBED / 4;
  cvt_f32_bf16<<<dim3((nx4 + 255) / 256), dim3(256), 0, stream>>>(x,  xb,  nx4);
  cvt_f32_bf16<<<dim3((nw4 + 255) / 256), dim3(256), 0, stream>>>(wq, wqb, nw4);
  cvt_f32_bf16<<<dim3((nw4 + 255) / 256), dim3(256), 0, stream>>>(wk, wkb, nw4);
  cvt_f32_bf16<<<dim3((nw4 + 255) / 256), dim3(256), 0, stream>>>(wv, wvb, nw4);
  cvt_f32_bf16<<<dim3((nw4 + 255) / 256), dim3(256), 0, stream>>>(wo, wob, nw4);

  dim3 gg(MTOT / 128, EMBED / 64);
  linear_wmma<0><<<gg, dim3(256), 0, stream>>>(xb, wqb, bq, qb);
  linear_wmma<0><<<gg, dim3(256), 0, stream>>>(xb, wkb, bk, kb);
  linear_wmma<1><<<gg, dim3(256), 0, stream>>>(xb, wvb, bv, vtb);

  flash_attn_wmma<<<dim3(SEQ / 128, BATCH * HEADS), dim3(256), 0, stream>>>(
      qb, kb, vtb, ab);

  linear_wmma<2><<<gg, dim3(256), 0, stream>>>(ab, wob, bo, d_out);
}
